// encoder_38276748542834
// MI455X (gfx1250) — matrix operand order repaired and checked by reading, not compiled, not run on hardware
//
#include <hip/hip_runtime.h>
#include <stdint.h>

typedef __attribute__((ext_vector_type(16))) _Float16 v16h;
typedef __attribute__((ext_vector_type(8)))  float    v8f;
typedef int v4i_vs __attribute__((__vector_size__(16)));   // matches builtin param type

union Frag16 { v16h v; uint4 q[2]; };

// Use CDNA5 async global->LDS copies when the toolchain exposes them.
#if defined(__HIP_DEVICE_COMPILE__) && \
    __has_builtin(__builtin_amdgcn_global_load_async_to_lds_b128) && \
    __has_builtin(__builtin_amdgcn_s_wait_asynccnt)
#define USE_ASYNC_LDS 1
#else
#define USE_ASYNC_LDS 0
#endif

#if USE_ASYNC_LDS
typedef __attribute__((address_space(1))) v4i_vs* async_gptr_t;
typedef __attribute__((address_space(3))) v4i_vs* async_lptr_t;
#endif

// Stage 16 bytes (8 halves) global -> LDS.
__device__ inline void stage16(const _Float16* g, _Float16* l) {
#if USE_ASYNC_LDS
  __builtin_amdgcn_global_load_async_to_lds_b128(
      (async_gptr_t)(v4i_vs*)const_cast<_Float16*>(g),
      (async_lptr_t)(v4i_vs*)l,
      0, 0);
#else
  *reinterpret_cast<uint4*>(l) = *reinterpret_cast<const uint4*>(g);
#endif
}

__device__ inline void stage_wait() {
#if USE_ASYNC_LDS
  __builtin_amdgcn_s_wait_asynccnt(0);
#endif
}

// ---------------------------------------------------------------------------
// WMMA fragment loaders per CDNA5 ISA 7.12.2 (wave32, 16-bit A 16x32, B 32x16)
// ---------------------------------------------------------------------------

// A[16][32] row-major (row stride lda halves).
// lanes 0-15: row=lane, elems 0..7 -> K 0..7, elems 8..15 -> K 16..23
// lanes 16-31: row=lane-16, K base shifted by +8
__device__ inline v16h load_frag_a(const _Float16* base, int lda) {
  int lane = threadIdx.x & 31;
  int row  = lane & 15;
  int kb   = (lane >> 4) << 3;           // 0 or 8
  const _Float16* p = base + (size_t)row * lda + kb;
  Frag16 f;
  f.q[0] = *reinterpret_cast<const uint4*>(p);       // K = kb+0..7
  f.q[1] = *reinterpret_cast<const uint4*>(p + 16);  // K = kb+16..23
  return f.v;
}

// B stored N-major: Bt[n][k] (row stride ldb halves), so a B column is a row.
// lanes 0-15: col=lane, elems 0..7 -> K 0..7, elems 8..15 -> K 16..23 ; lanes 16-31: col=lane-16, K base +8
__device__ inline v16h load_frag_b(const _Float16* base, int ldb) {
  int lane = threadIdx.x & 31;
  int col  = lane & 15;
  int kb   = (lane >> 4) << 3;           // 0 or 8
  const _Float16* p = base + (size_t)col * ldb + kb;
  Frag16 f;
  f.q[0] = *reinterpret_cast<const uint4*>(p);       // K = kb+0..7
  f.q[1] = *reinterpret_cast<const uint4*>(p + 16);  // K = kb+16..23
  return f.v;
}

__device__ inline v8f wmma_f16(v16h a, v16h b, v8f c) {
  return __builtin_amdgcn_wmma_f32_16x16x32_f16(false, a, false, b, (short)0, c,
                                                false, false);
}

// ---------------------------------------------------------------------------
// Generic f16 WMMA GEMM: C[M][N] = A[M][K] * Bt[N][K]^T (+bias)(+residual)(relu)
// Block tile 128(M) x 256(N), 8 waves (2x4), wave tile 64x64 (16 WMMA accums),
// K staged 32-wide in LDS via async global->LDS copies.
// Per K-step per wave: 16 ds_load_b128 -> 16 v_wmma (1:1).
// ---------------------------------------------------------------------------
__global__ __launch_bounds__(256) void wmma_gemm_kernel(
    const _Float16* __restrict__ A,        // [M][K]
    const _Float16* __restrict__ Bt,       // [N][K]  (N-major weight)
    const float*    __restrict__ bias,     // [N] or null
    const float*    __restrict__ residual, // [M][N] or null
    float*          __restrict__ Cf,       // f32 out or null
    _Float16*       __restrict__ Ch,       // f16 out or null
    int M, int N, int K, int relu)
{
  __shared__ _Float16 As[128][40];   // pad: 80B row stride (16B aligned, conflict-free)
  __shared__ _Float16 Bs[256][40];

  const int m0   = blockIdx.x * 128;
  const int n0   = blockIdx.y * 256;
  const int tid  = threadIdx.x;
  const int wave = tid >> 5;
  const int wm   = wave >> 2;        // 0..1  (64 rows each)
  const int wn   = wave & 3;         // 0..3  (64 cols each)

  v8f acc[4][4];
  for (int i = 0; i < 4; i++)
    for (int j = 0; j < 4; j++)
      acc[i][j] = (v8f){0.f,0.f,0.f,0.f,0.f,0.f,0.f,0.f};

  const int arow = tid >> 1;          // 0..127
  const int akoff = (tid & 1) * 16;   // 0 or 16

  for (int kk = 0; kk < K; kk += 32) {
    __syncthreads();   // previous iteration's LDS reads complete
    {
      const _Float16* ga = A + (size_t)(m0 + arow) * K + kk + akoff;
      stage16(ga,     &As[arow][akoff]);
      stage16(ga + 8, &As[arow][akoff + 8]);
      const _Float16* gb = Bt + (size_t)(n0 + tid) * K + kk;
      stage16(gb,      &Bs[tid][0]);
      stage16(gb + 8,  &Bs[tid][8]);
      stage16(gb + 16, &Bs[tid][16]);
      stage16(gb + 24, &Bs[tid][24]);
    }
    stage_wait();
    __syncthreads();

    v16h bf[4];
#pragma unroll
    for (int ni = 0; ni < 4; ni++)
      bf[ni] = load_frag_b(&Bs[wn * 64 + ni * 16][0], 40);
#pragma unroll
    for (int mi = 0; mi < 4; mi++) {
      v16h af = load_frag_a(&As[wm * 64 + mi * 16][0], 40);
#pragma unroll
      for (int ni = 0; ni < 4; ni++)
        acc[mi][ni] = wmma_f16(af, bf[ni], acc[mi][ni]);
    }
  }

  // Epilogue. C/D layout: lane<16 -> N=lane, VGPR j -> M=j ; lane>=16 -> N=lane-16, M=8+j
  const int lane  = tid & 31;
  const int ncol  = lane & 15;
  const int mbase = (lane >> 4) * 8;
#pragma unroll
  for (int mi = 0; mi < 4; mi++) {
#pragma unroll
    for (int ni = 0; ni < 4; ni++) {
      const int gn = n0 + wn * 64 + ni * 16 + ncol;
      const float bv = bias ? bias[gn] : 0.0f;
#pragma unroll
      for (int j = 0; j < 8; j++) {
        const int gm = m0 + wm * 64 + mi * 16 + mbase + j;
        const size_t off = (size_t)gm * N + gn;
        float v = acc[mi][ni][j] + bv;
        if (residual) v += residual[off];
        if (relu) v = fmaxf(v, 0.0f);
        if (Cf) Cf[off] = v;
        if (Ch) Ch[off] = (_Float16)v;
      }
    }
  }
}

// ---------------------------------------------------------------------------
// Autocorrelation via Gram diagonals:
//   mean_value[b,tau] = (1/D) * sum_s <Q[b,s,:], K[b,(s-tau)%L,:]>
// Each wave computes one 16x16 tile of G=Q*K^T (K=512) straight from global
// (f16 rows are contiguous; L2 holds Q/K entirely), then scatter-adds the
// wrapped diagonals into mean_value.
// ---------------------------------------------------------------------------
__global__ __launch_bounds__(256) void corr_diag_kernel(
    const _Float16* __restrict__ Q, const _Float16* __restrict__ Km,
    float* __restrict__ meanv, int L, int D)
{
  const int b  = blockIdx.z;
  const int s0 = blockIdx.x * 16;
  const int j0 = (blockIdx.y * 8 + (threadIdx.x >> 5)) * 16;
  const _Float16* Qb = Q  + (size_t)b * L * D;
  const _Float16* Kb = Km + (size_t)b * L * D;

  v8f acc = (v8f){0.f,0.f,0.f,0.f,0.f,0.f,0.f,0.f};
  for (int k0 = 0; k0 < D; k0 += 32) {
    v16h a  = load_frag_a(Qb + (size_t)s0 * D + k0, D);
    v16h bf = load_frag_b(Kb + (size_t)j0 * D + k0, D);  // Bt[n][k] = K row (j0+n)
    acc = wmma_f16(a, bf, acc);
  }

  const int lane  = threadIdx.x & 31;
  const int ncol  = lane & 15;
  const int mbase = (lane >> 4) * 8;
  const float scale = 1.0f / (float)D;
#pragma unroll
  for (int j = 0; j < 8; j++) {
    const int tau = (s0 + mbase + j - j0 - ncol + L) & (L - 1);  // L = 4096 (pow2)
    atomicAdd(&meanv[(size_t)b * L + tau], acc[j] * scale);
  }
}

// ---------------------------------------------------------------------------
// Small helper kernels
// ---------------------------------------------------------------------------
__global__ void zero_f32_kernel(float* __restrict__ p, size_t n) {
  size_t e = (size_t)blockIdx.x * blockDim.x + threadIdx.x;
  if (e < n) p[e] = 0.0f;
}

__global__ void f32_to_f16_kernel(const float* __restrict__ in,
                                  _Float16* __restrict__ out, size_t n) {
  size_t e = (size_t)blockIdx.x * blockDim.x + threadIdx.x;
  if (e < n) out[e] = (_Float16)in[e];
}

// Wt[n][k] = (f16) W[k][n]   (weights are 512x512; cache-friendly enough naive)
__global__ void transpose_f16_kernel(const float* __restrict__ W,
                                     _Float16* __restrict__ Wt, int K, int N) {
  size_t e = (size_t)blockIdx.x * blockDim.x + threadIdx.x;
  if (e >= (size_t)K * N) return;
  int k = (int)(e % K);
  int n = (int)(e / K);
  Wt[e] = (_Float16)W[(size_t)k * N + n];
}

// top-8 of batch-mean mean_value + per-batch softmax weights (single block)
__global__ __launch_bounds__(256) void topk_kernel(
    const float* __restrict__ meanv, int* __restrict__ idx,
    float* __restrict__ w, int B, int L, int k)
{
  __shared__ float avg[4096];
  __shared__ float red[256];
  __shared__ int   redi[256];
  __shared__ int   sIdx[8];
  const int tid = threadIdx.x;

  for (int t = tid; t < L; t += 256) {
    float s = 0.0f;
    for (int b = 0; b < B; b++) s += meanv[(size_t)b * L + t];
    avg[t] = s / (float)B;
  }
  __syncthreads();

  for (int i = 0; i < k; i++) {
    float best = -3.4e38f; int bi = 0;
    for (int t = tid; t < L; t += 256)
      if (avg[t] > best) { best = avg[t]; bi = t; }
    red[tid] = best; redi[tid] = bi;
    __syncthreads();
    for (int s = 128; s > 0; s >>= 1) {
      if (tid < s && red[tid + s] > red[tid]) {
        red[tid] = red[tid + s]; redi[tid] = redi[tid + s];
      }
      __syncthreads();
    }
    if (tid == 0) { sIdx[i] = redi[0]; idx[i] = redi[0]; avg[redi[0]] = -3.4e38f; }
    __syncthreads();
  }

  if (tid < B) {
    const int b = tid;
    float vals[8];
    float m = -3.4e38f;
    for (int i = 0; i < k; i++) {
      vals[i] = meanv[(size_t)b * L + sIdx[i]];
      m = fmaxf(m, vals[i]);
    }
    float sum = 0.0f;
    for (int i = 0; i < k; i++) { vals[i] = __expf(vals[i] - m); sum += vals[i]; }
    for (int i = 0; i < k; i++) w[b * k + i] = vals[i] / sum;
  }
}

// agg[b,t,:] = sum_i w[b,i] * v[b,(t+idx[i])%L,:]
__global__ void agg_kernel(const _Float16* __restrict__ V, const int* __restrict__ idx,
                           const float* __restrict__ w, _Float16* __restrict__ out,
                           int B, int L, int D, int k)
{
  size_t e = (size_t)blockIdx.x * blockDim.x + threadIdx.x;
  if (e >= (size_t)B * L * D) return;
  const int d = (int)(e % D);
  const int t = (int)((e / D) % L);
  const int b = (int)(e / ((size_t)L * D));
  float acc = 0.0f;
  for (int i = 0; i < k; i++) {
    const int src = (t + idx[i]) & (L - 1);
    acc += (float)V[((size_t)b * L + src) * D + d] * w[b * k + i];
  }
  out[e] = (_Float16)acc;
}

// out = X - moving_avg(X, 2*hw+1)  (TF SAME, padding excluded from count)
__global__ void decomp_kernel(const float* __restrict__ X, float* __restrict__ outf,
                              _Float16* __restrict__ outh, int B, int L, int D, int hw)
{
  size_t e = (size_t)blockIdx.x * blockDim.x + threadIdx.x;
  if (e >= (size_t)B * L * D) return;
  const int d = (int)(e % D);
  const int t = (int)((e / D) % L);
  const int b = (int)(e / ((size_t)L * D));
  const int lo = (t - hw > 0) ? (t - hw) : 0;
  const int hi = (t + hw < L - 1) ? (t + hw) : (L - 1);
  const float* col = X + (size_t)b * L * D + d;
  float s = 0.0f;
  for (int u = lo; u <= hi; u++) s += col[(size_t)u * D];
  const float v = X[e] - s / (float)(hi - lo + 1);
  if (outf) outf[e] = v;
  if (outh) outh[e] = (_Float16)v;
}

// ---------------------------------------------------------------------------
// Host orchestration
// ---------------------------------------------------------------------------
extern "C" void kernel_launch(void* const* d_in, const int* in_sizes, int n_in,
                              void* d_out, int out_size, void* d_ws, size_t ws_size,
                              hipStream_t stream)
{
  (void)in_sizes; (void)n_in; (void)out_size; (void)ws_size;
  const int B = 4, L = 4096, D = 512;
  const int TOPK = 8;    // int(c * ln(L)) = int(8.317)
  const int HW   = 12;   // (kernel_size-1)/2 = 12
  const size_t ML    = (size_t)B * L;   // 16384 rows
  const size_t NElem = ML * D;          // 8388608

  const float* x  = (const float*)d_in[0];
  const float* bi = (const float*)d_in[2];
  const float* bq = (const float*)d_in[4];
  const float* bk = (const float*)d_in[6];
  const float* bv = (const float*)d_in[8];
  const float* bo = (const float*)d_in[10];
  float* out = (float*)d_out;

  char* ws = (char*)d_ws;
  size_t off = 0;
  auto alloc = [&](size_t bytes) -> void* {
    void* p = ws + off;
    off = (off + bytes + 255) & ~(size_t)255;
    return p;
  };
  _Float16* x16 = (_Float16*)alloc(NElem * 2);   // later reused as agg16
  _Float16* h16 = (_Float16*)alloc(NElem * 2);   // later reused as s16
  _Float16* q16 = (_Float16*)alloc(NElem * 2);   // later reused as y1_16
  _Float16* k16 = (_Float16*)alloc(NElem * 2);
  _Float16* v16 = (_Float16*)alloc(NElem * 2);
  float*    r32 = (float*)alloc(NElem * 4);      // later reused as y2
  _Float16* Wt[7];
  for (int i = 0; i < 7; i++) Wt[i] = (_Float16*)alloc((size_t)D * D * 2);
  float* meanv = (float*)alloc((size_t)B * L * 4);
  int*   idx   = (int*)alloc(TOPK * 4);
  float* wts   = (float*)alloc((size_t)B * TOPK * 4);

  _Float16* agg16 = x16;   // x16 dead after GEMM1 (residual uses f32 x)
  _Float16* s16   = h16;   // h16 dead after q/k/v GEMMs
  _Float16* y1_16 = q16;   // q16 dead after correlation
  float*    y2    = r32;   // r32 dead after seasonal decomposition

  const unsigned g256 = (unsigned)((NElem + 255) / 256);

  // Prep: x -> f16, weights -> transposed f16, zero mean_value accumulator
  f32_to_f16_kernel<<<g256, 256, 0, stream>>>(x, x16, NElem);
  const float* Wsrc[7] = { (const float*)d_in[1], (const float*)d_in[3],
                           (const float*)d_in[5], (const float*)d_in[7],
                           (const float*)d_in[9], (const float*)d_in[11],
                           (const float*)d_in[12] };
  for (int i = 0; i < 7; i++)
    transpose_f16_kernel<<<(D * D + 255) / 256, 256, 0, stream>>>(Wsrc[i], Wt[i], D, D);
  zero_f32_kernel<<<(B * L + 255) / 256, 256, 0, stream>>>(meanv, (size_t)B * L);

  const dim3 gGemm((unsigned)(ML / 128), (unsigned)(D / 256));
  // h = x@Wi + bi (f16) ; q,k,v = h@W* + b* (f16)
  wmma_gemm_kernel<<<gGemm, 256, 0, stream>>>(x16, Wt[0], bi, nullptr, nullptr, h16, (int)ML, D, D, 0);
  wmma_gemm_kernel<<<gGemm, 256, 0, stream>>>(h16, Wt[1], bq, nullptr, nullptr, q16, (int)ML, D, D, 0);
  wmma_gemm_kernel<<<gGemm, 256, 0, stream>>>(h16, Wt[2], bk, nullptr, nullptr, k16, (int)ML, D, D, 0);
  wmma_gemm_kernel<<<gGemm, 256, 0, stream>>>(h16, Wt[3], bv, nullptr, nullptr, v16, (int)ML, D, D, 0);

  // mean_value via Gram-matrix wrapped diagonals (all-WMMA)
  const dim3 gCorr((unsigned)(L / 16), (unsigned)(L / 16 / 8), (unsigned)B);
  corr_diag_kernel<<<gCorr, 256, 0, stream>>>(q16, k16, meanv, L, D);

  topk_kernel<<<1, 256, 0, stream>>>(meanv, idx, wts, B, L, TOPK);
  agg_kernel<<<g256, 256, 0, stream>>>(v16, idx, wts, agg16, B, L, D, TOPK);

  // r = agg@Wo + bo + x (f32) ; s = r - mavg(r) (f16)
  wmma_gemm_kernel<<<gGemm, 256, 0, stream>>>(agg16, Wt[4], bo, x, r32, nullptr, (int)ML, D, D, 0);
  decomp_kernel<<<g256, 256, 0, stream>>>(r32, nullptr, s16, B, L, D, HW);

  // y = relu(s@W1)@W2 ; out = y - mavg(y)
  wmma_gemm_kernel<<<gGemm, 256, 0, stream>>>(s16, Wt[5], nullptr, nullptr, nullptr, y1_16, (int)ML, D, D, 1);
  wmma_gemm_kernel<<<gGemm, 256, 0, stream>>>(y1_16, Wt[6], nullptr, nullptr, y2, nullptr, (int)ML, D, D, 0);
  decomp_kernel<<<g256, 256, 0, stream>>>(y2, out, nullptr, B, L, D, HW);
}
